// LMModel_LSTM_55894704390353
// MI455X (gfx1250) — matrix operand order repaired and checked by reading, not compiled, not run on hardware
//
#include <hip/hip_runtime.h>
#include <hip/hip_bf16.h>
#include <cstdint>

typedef __attribute__((ext_vector_type(16))) _Float16 v16h;
typedef __attribute__((ext_vector_type(8)))  _Float16 v8h;
typedef __attribute__((ext_vector_type(8)))  float    v8f;

#define NVOC  50257
#define DIM   256
#define HID   256
#define SEQ   128
#define BATCH 16
#define TB    (SEQ*BATCH)   // 2048 rows of h_top
#define NPAD  50304         // 393 * 128, padded vocab for tiling

// padded LDS row strides (avoid 64-bank conflicts: stride % 64 dwords == 4)
#define ACT_STRIDE 520      // halves per act row  (1040 B)
#define AS_STRIDE  264      // halves per As row   (528 B)
#define PRE_STRIDE 260      // floats per pre row  (1040 B)

// ---- workspace layout (byte offsets, all 256B aligned) ----
#define WS_WL     0u                        // gate weights, WMMA-B frag layout: 4 WGs * 512*256 f16 = 1,048,576 B
#define WS_XSEQ   (WS_WL    + 1048576u)     // embedded inputs f16: 2048*256        = 1,048,576 B
#define WS_HTOP   (WS_XSEQ  + 1048576u)     // top-layer h f16:     2048*256        = 1,048,576 B
#define WS_DECW   (WS_HTOP  + 1048576u)     // dec_w f16 padded:    50304*256       = 25,755,648 B
#define WS_BCOMB  (WS_DECW  + 25755648u)    // combined bias f32:   4*256           = 4,096 B
#define WS_HBUF   (WS_BCOMB + 4096u)        // h state f16: [layer][parity][16][256]= 32,768 B
#define WS_CNT    (WS_HBUF  + 32768u)       // barrier counter

__device__ __forceinline__ float sigm(float x) { return 1.0f / (1.0f + __expf(-x)); }

// ---------- CDNA5 async global->LDS copy (ASYNCcnt path), with safe fallback ----------
#if defined(__HIP_DEVICE_COMPILE__) && \
    __has_builtin(__builtin_amdgcn_global_load_async_to_lds_b128) && \
    __has_builtin(__builtin_amdgcn_s_wait_asynccnt)
#define ASYNC_LDS 1
typedef int i4v __attribute__((vector_size(16)));           // matches builtin's V4i
typedef __attribute__((address_space(1))) i4v GI4;          // global (prints as __device__)
typedef __attribute__((address_space(3))) i4v LI4;          // LDS
__device__ __forceinline__ void async_b128(void* lds, const void* g) {
  __builtin_amdgcn_global_load_async_to_lds_b128(
      (GI4*)(uintptr_t)g, (LI4*)(unsigned)(uintptr_t)lds, 0, 0);
}
__device__ __forceinline__ void async_wait0() { __builtin_amdgcn_s_wait_asynccnt(0); }
#endif

__device__ __forceinline__ void copy64B_to_lds(void* lds, const void* g) {
#ifdef ASYNC_LDS
  async_b128((char*)lds,      (const char*)g);
  async_b128((char*)lds + 16, (const char*)g + 16);
  async_b128((char*)lds + 32, (const char*)g + 32);
  async_b128((char*)lds + 48, (const char*)g + 48);
#else
  const uint4* s4 = (const uint4*)g;
  uint4* d4 = (uint4*)lds;
  uint4 t0 = s4[0], t1 = s4[1], t2 = s4[2], t3 = s4[3];
  d4[0] = t0; d4[1] = t1; d4[2] = t2; d4[3] = t3;
#endif
}
__device__ __forceinline__ void copy_fence() {
#ifdef ASYNC_LDS
  async_wait0();
#endif
}

// ---------------- init: zero barrier counter + h state, build combined bias ----------------
__global__ void k_init(unsigned* cnt, unsigned* hbuf32, float* bcomb,
                       const float* bf, const float* bi, const float* bc, const float* bo) {
  int tid = threadIdx.x;
  if (tid == 0) *cnt = 0u;
  for (int i = tid; i < 8192; i += 256) hbuf32[i] = 0u;     // h0 = 0 (f16 pairs)
  for (int i = tid; i < 1024; i += 256) {                   // bcomb[wg*256 + nloc]
    int wg = i >> 8, nloc = i & 255, G = nloc >> 6, j = nloc & 63;
    const float* b = (G == 0) ? bf : (G == 1) ? bi : (G == 2) ? bo : bc;
    bcomb[i] = b[wg * 64 + j];
  }
}

// ---------------- embedding gather -> f16 row-major [t*16+b][256] ----------------
__global__ void k_embed(const int* __restrict__ tokens, const float* __restrict__ emb,
                        _Float16* __restrict__ xseq) {
  int row = blockIdx.x;            // 0..2047 = t*16 + b
  int k   = threadIdx.x;           // 0..255
  int tok = tokens[row];
  xseq[row * 256 + k] = (_Float16)emb[(size_t)tok * 256 + k];
}

// ---------------- gate weights -> f16, WMMA B-fragment order per owning WG ----------------
// wl[idx]: idx = wg*131072 + (kt*16+nt)*512 + lane*16 + half
// B element (k,n) in a 32x16 K-tile: lane = ((k/16)&1)*16 + n%16, half = k%16
__global__ void k_prep_w(const float* __restrict__ wf, const float* __restrict__ uf,
                         const float* __restrict__ wi, const float* __restrict__ ui,
                         const float* __restrict__ wo, const float* __restrict__ uo,
                         const float* __restrict__ wc, const float* __restrict__ uc,
                         _Float16* __restrict__ wl) {
  int idx  = blockIdx.x * 256 + threadIdx.x;  // 0 .. 524287
  int wg   = idx >> 17;
  int rem  = idx & 131071;
  int frag = rem >> 9;
  int lane = (rem >> 4) & 31;
  int half = rem & 15;
  int kt = frag >> 4, nt = frag & 15;
  int k    = kt * 32 + (lane >> 4) * 16 + half;      // 0..511 (rows: [W ; U])
  int nloc = nt * 16 + (lane & 15);                  // 0..255 local gate-col
  int G = nloc >> 6, j = nloc & 63;
  int col = wg * 64 + j;                             // hidden unit
  const float* W = (G == 0) ? wf : (G == 1) ? wi : (G == 2) ? wo : wc;
  const float* U = (G == 0) ? uf : (G == 1) ? ui : (G == 2) ? uo : uc;
  float v = (k < 256) ? W[k * 256 + col] : U[(k - 256) * 256 + col];
  wl[idx] = (_Float16)v;
}

// ---------------- dec_w -> f16, padded to NPAD rows ----------------
__global__ void k_prep_decw(const float* __restrict__ dec_w, _Float16* __restrict__ w16) {
  int n = blockIdx.x;              // 0..NPAD-1
  int k = threadIdx.x;             // 0..255
  float v = (n < NVOC) ? dec_w[(size_t)n * 256 + k] : 0.0f;
  w16[(size_t)n * 256 + k] = (_Float16)v;
}

// ---------------- persistent LSTM recurrence: 4 WGs, weights resident in LDS ----------------
__global__ void __launch_bounds__(256) k_lstm(
    const _Float16* __restrict__ wl_g, const float* __restrict__ bcomb,
    const _Float16* __restrict__ xseq, _Float16* __restrict__ hbuf,
    _Float16* __restrict__ htop, float* __restrict__ outHT, float* __restrict__ outCT,
    unsigned* cnt) {
  extern __shared__ char smem_raw[];
  _Float16* wl  = (_Float16*)smem_raw;             // 131072 halves = 256 KB weight slice
  _Float16* act = wl + 131072;                     // [16][ACT_STRIDE] f16 (A = [h_prev ; x])
  float*    pre = (float*)(act + 16 * ACT_STRIDE); // [16][PRE_STRIDE] f32
  float*    cst = pre + 16 * PRE_STRIDE;           // [2][16][64] f32
  float*    bl  = cst + 2048;                      // [256] f32

  const int tid  = threadIdx.x;
  const int wg   = blockIdx.x;            // owns hidden units [64*wg, 64*wg+64)
  const int lane = tid & 31;
  const int wv   = tid >> 5;              // 8 waves
  const int lc   = lane & 15;             // column-in-tile / row-in-tile
  const int grp  = lane >> 4;

  // preload weight slice into LDS (stays resident for entire recurrence)
  {
    const char* src = (const char*)wl_g + (size_t)wg * 262144;
    for (int i = tid * 64; i < 262144; i += 256 * 64)
      copy64B_to_lds((char*)wl + i, src + i);
    copy_fence();
  }
  bl[tid] = bcomb[wg * 256 + tid];
  for (int i = tid; i < 2048; i += 256) cst[i] = 0.0f;      // c0 = 0
  __syncthreads();

  unsigned gen = 0;
  for (int t = 0; t < SEQ; ++t) {
    const int rp = t & 1, wp = (t + 1) & 1;                 // double-buffered h parity
    for (int l = 0; l < 2; ++l) {
      // ---- stage A = [h_prev(l) ; x] into LDS ----
      {
        int b = tid >> 4, seg = tid & 15;                   // seg: 32-half chunk
        const _Float16* src;
        if (seg < 8)          src = hbuf + ((l * 2 + rp) * 16 + b) * 256 + seg * 32;
        else if (l == 0)      src = xseq + (t * 16 + b) * 256 + (seg - 8) * 32;
        else                  src = hbuf + ((0 * 2 + wp) * 16 + b) * 256 + (seg - 8) * 32;
        copy64B_to_lds(act + b * ACT_STRIDE + seg * 32, src);
        copy_fence();
      }
      __syncthreads();

      // ---- gate GEMM: (16 x 512) @ (512 x 256 local); A shared by 2 N-tiles ----
      const int nt0 = wv * 2, nt1 = nt0 + 1;
      const _Float16* abase = act + lc * ACT_STRIDE + grp * 16;
      v8f acc0 = {}, acc1 = {};
      #pragma unroll
      for (int kt = 0; kt < 16; ++kt) {
        union { v16h v; v8h h[2]; } a;
        const _Float16* ap = abase + kt * 32;
        a.h[0] = *(const v8h*)ap;
        a.h[1] = *(const v8h*)(ap + 8);
        v16h b0 = *(const v16h*)(wl + ((kt * 16 + nt0) << 9) + lane * 16);
        v16h b1 = *(const v16h*)(wl + ((kt * 16 + nt1) << 9) + lane * 16);
        acc0 = __builtin_amdgcn_wmma_f32_16x16x32_f16(false, a.v, false, b0,
                                                      (short)0, acc0, false, false);
        acc1 = __builtin_amdgcn_wmma_f32_16x16x32_f16(false, a.v, false, b1,
                                                      (short)0, acc1, false, false);
      }
      const float bias0 = bl[nt0 * 16 + lc];
      const float bias1 = bl[nt1 * 16 + lc];
      #pragma unroll
      for (int r = 0; r < 8; ++r) {
        pre[(r + grp * 8) * PRE_STRIDE + nt0 * 16 + lc] = acc0[r] + bias0;
        pre[(r + grp * 8) * PRE_STRIDE + nt1 * 16 + lc] = acc1[r] + bias1;
      }
      __syncthreads();

      // ---- elementwise LSTM cell update for owned hidden units ----
      #pragma unroll
      for (int e = 0; e < 4; ++e) {
        int idx = (tid << 2) + e;                           // 0..1023
        int b = idx >> 6, j = idx & 63;
        float f_ = sigm (pre[b * PRE_STRIDE +   0 + j]);
        float i_ = sigm (pre[b * PRE_STRIDE +  64 + j]);
        float o_ = sigm (pre[b * PRE_STRIDE + 128 + j]);
        float g_ = tanhf(pre[b * PRE_STRIDE + 192 + j]);
        int ci = (l * 16 + b) * 64 + j;
        float cp = cst[ci];
        float cn = f_ * cp + i_ * g_;
        cst[ci] = cn;
        float hn = o_ * tanhf(cn);
        hbuf[((l * 2 + wp) * 16 + b) * 256 + wg * 64 + j] = (_Float16)hn;
        if (l == 1) htop[(t * 16 + b) * 256 + wg * 64 + j] = (_Float16)hn;
        if (t == SEQ - 1) {
          outHT[(l * 16 + b) * 256 + wg * 64 + j] = hn;
          outCT[(l * 16 + b) * 256 + wg * 64 + j] = cn;
        }
      }

      // ---- device-wide sense barrier (4 resident WGs) ----
      ++gen;
      __syncthreads();
      if (tid == 0) {
        __threadfence();
        atomicAdd(cnt, 1u);
        while (atomicAdd(cnt, 0u) < 4u * gen) __builtin_amdgcn_s_sleep(2);
      }
      __syncthreads();
    }
  }
}

// ---------------- decoder GEMM: (2048 x 256) @ (256 x NVOC) + bias ----------------
// WG = 32 M-rows x 128 N-cols; 8 waves; each wave: 1 N-tile x 2 M-tiles sharing B.
__global__ void __launch_bounds__(256) k_decode(
    const _Float16* __restrict__ htop, const _Float16* __restrict__ w16,
    const float* __restrict__ dec_b, float* __restrict__ out) {
  __shared__ _Float16 As[32 * AS_STRIDE];
  const int tid = threadIdx.x, lane = tid & 31, wv = tid >> 5;
  const int lc = lane & 15, grp = lane >> 4;
  {
    int row = tid >> 3, seg = tid & 7;
    copy64B_to_lds(As + row * AS_STRIDE + seg * 32,
                   htop + (size_t)(blockIdx.y * 32 + row) * 256 + seg * 32);
    copy_fence();
  }
  __syncthreads();

  const int n = blockIdx.x * 128 + wv * 16 + lc;            // output vocab column
  const _Float16* bbase = w16 + (size_t)n * 256 + grp * 16;
  const _Float16* a0base = As + lc * AS_STRIDE + grp * 16;
  const _Float16* a1base = a0base + 16 * AS_STRIDE;

  v8f acc0 = {}, acc1 = {};
  #pragma unroll
  for (int kt = 0; kt < 8; ++kt) {
    v16h b = *(const v16h*)(bbase + kt * 32);
    if (kt < 7) __builtin_prefetch(bbase + kt * 32 + 32, 0, 1);
    union { v16h v; v8h h[2]; } a0, a1;
    a0.h[0] = *(const v8h*)(a0base + kt * 32);
    a0.h[1] = *(const v8h*)(a0base + kt * 32 + 8);
    a1.h[0] = *(const v8h*)(a1base + kt * 32);
    a1.h[1] = *(const v8h*)(a1base + kt * 32 + 8);
    acc0 = __builtin_amdgcn_wmma_f32_16x16x32_f16(false, a0.v, false, b,
                                                  (short)0, acc0, false, false);
    acc1 = __builtin_amdgcn_wmma_f32_16x16x32_f16(false, a1.v, false, b,
                                                  (short)0, acc1, false, false);
  }
  if (n < NVOC) {
    const float bias = dec_b[n];
    const int rb = blockIdx.y * 32 + grp * 8;               // C: M = r + grp*8
    #pragma unroll
    for (int r = 0; r < 8; ++r) {
      out[(size_t)(rb + r) * NVOC + n]      = acc0[r] + bias;
      out[(size_t)(rb + 16 + r) * NVOC + n] = acc1[r] + bias;
    }
  }
}

extern "C" void kernel_launch(void* const* d_in, const int* in_sizes, int n_in,
                              void* d_out, int out_size, void* d_ws, size_t ws_size,
                              hipStream_t stream) {
  (void)in_sizes; (void)n_in; (void)out_size; (void)ws_size;
  const int*   tokens = (const int*)d_in[0];
  const float* emb    = (const float*)d_in[1];
  const float* wf = (const float*)d_in[2];
  const float* uf = (const float*)d_in[3];
  const float* bf = (const float*)d_in[4];
  const float* wi = (const float*)d_in[5];
  const float* ui = (const float*)d_in[6];
  const float* bi = (const float*)d_in[7];
  const float* wc = (const float*)d_in[8];
  const float* uc = (const float*)d_in[9];
  const float* bc = (const float*)d_in[10];
  const float* wo = (const float*)d_in[11];
  const float* uo = (const float*)d_in[12];
  const float* bo = (const float*)d_in[13];
  const float* dec_w = (const float*)d_in[14];
  const float* dec_b = (const float*)d_in[15];

  char* ws = (char*)d_ws;
  _Float16* wl16   = (_Float16*)(ws + WS_WL);
  _Float16* xseq   = (_Float16*)(ws + WS_XSEQ);
  _Float16* htop   = (_Float16*)(ws + WS_HTOP);
  _Float16* decw16 = (_Float16*)(ws + WS_DECW);
  float*    bcomb  = (float*)   (ws + WS_BCOMB);
  _Float16* hbuf   = (_Float16*)(ws + WS_HBUF);
  unsigned* cnt    = (unsigned*)(ws + WS_CNT);

  float* out   = (float*)d_out;
  float* outHT = out + (size_t)TB * NVOC;       // logits first, then hT, then cT
  float* outCT = outHT + 2 * 16 * 256;

  const int LSTM_SMEM = 262144 + 16 * ACT_STRIDE * 2 + 16 * PRE_STRIDE * 4 + 8192 + 1024;
  (void)hipFuncSetAttribute((const void*)k_lstm,
                            hipFuncAttributeMaxDynamicSharedMemorySize, LSTM_SMEM);

  k_init     <<<1,    256, 0, stream>>>(cnt, (unsigned*)hbuf, bcomb, bf, bi, bc, bo);
  k_embed    <<<TB,   256, 0, stream>>>(tokens, emb, xseq);
  k_prep_w   <<<2048, 256, 0, stream>>>(wf, uf, wi, ui, wo, uo, wc, uc, wl16);
  k_prep_decw<<<NPAD, 256, 0, stream>>>(dec_w, decw16);
  k_lstm     <<<4,    256, LSTM_SMEM, stream>>>(wl16, bcomb, xseq, hbuf, htop, outHT, outCT, cnt);
  dim3 dg(NPAD / 128, TB / 32);                 // 393 x 64 workgroups
  k_decode   <<<dg,   256, 0, stream>>>(htop, decw16, dec_b, out);
}
